// SiameseNetworkRespective_12335146074640
// MI455X (gfx1250) — matrix operand order repaired and checked by reading, not compiled, not run on hardware
//
#include <hip/hip_runtime.h>
#include <hip/hip_bf16.h>

typedef __attribute__((ext_vector_type(16))) _Float16 v16h;
typedef __attribute__((ext_vector_type(8)))  _Float16 v8h;
typedef __attribute__((ext_vector_type(8)))  float    v8f;
typedef __attribute__((ext_vector_type(4)))  float    v4f;

#define INC  64
#define OUTC 128

// ---------------------------------------------------------------------------
// WMMA helper: D = A(16x32 f16) * B(32x16 f16) + C(16x16 f32)
// ---------------------------------------------------------------------------
__device__ __forceinline__ v8f wmma_16x16x32(v16h a, v16h b, v8f c) {
  return __builtin_amdgcn_wmma_f32_16x16x32_f16(
      /*neg_a=*/false, a, /*neg_b=*/false, b,
      /*c_mod=*/(short)0, c, /*reuse_a=*/false, /*reuse_b=*/false);
}

// A-tile (16x32, f16) load from row-major f32, ISA layout:
//   row = lane%16 ; halves 0..7 -> K = 8*grp + 0..7 ; halves 8..15 -> K = 16 + 8*grp + 0..7
// p0 points at src[row*stride + ktile*32 + 8*grp]
__device__ __forceinline__ v16h loadA_global_f32(const float* p0) {
  const v4f* q = (const v4f*)p0;
  v4f f0 = q[0], f1 = q[1];   // K offsets +0..7
  v4f f2 = q[4], f3 = q[5];   // K offsets +16..23
  v16h a;
  a[0]=(_Float16)f0.x; a[1]=(_Float16)f0.y; a[2]=(_Float16)f0.z; a[3]=(_Float16)f0.w;
  a[4]=(_Float16)f1.x; a[5]=(_Float16)f1.y; a[6]=(_Float16)f1.z; a[7]=(_Float16)f1.w;
  a[8]=(_Float16)f2.x; a[9]=(_Float16)f2.y; a[10]=(_Float16)f2.z; a[11]=(_Float16)f2.w;
  a[12]=(_Float16)f3.x; a[13]=(_Float16)f3.y; a[14]=(_Float16)f3.z; a[15]=(_Float16)f3.w;
  return a;
}

// Same A layout but from an f16 LDS slice (row stride 128 halves).
__device__ __forceinline__ v16h loadA_lds_f16(const _Float16* p0) {
  v8h x0 = *(const v8h*)(p0);        // ds_load_b128
  v8h x1 = *(const v8h*)(p0 + 16);   // ds_load_b128
  v16h a;
#pragma unroll
  for (int i = 0; i < 8; ++i) { a[i] = x0[i]; a[i + 8] = x1[i]; }
  return a;
}

// ---------------------------------------------------------------------------
// Weight pre-swizzle: W (row-major [128][K]) -> B-tile lane layout, f16.
//   swz[((tk*8+tn)*32+lane)*16 + h] = W[tn*16 + lane%16][tk*32 + 8*(lane/16) + (h%8) + 16*(h/8)]
// ---------------------------------------------------------------------------
__global__ void k_swz(const float* __restrict__ W, _Float16* __restrict__ swz, int K) {
  int idx = blockIdx.x * blockDim.x + threadIdx.x;
  int total = (K / 32) * 8 * 32 * 16;
  if (idx >= total) return;
  int h    = idx & 15;
  int lane = (idx >> 4) & 31;
  int tn   = (idx >> 9) & 7;
  int tk   = idx >> 12;
  int o = tn * 16 + (lane & 15);
  int k = tk * 32 + 8 * (lane >> 4) + (h & 7) + 16 * (h >> 3);
  swz[idx] = (_Float16)W[(size_t)o * K + k];
}

__global__ void k_zero(float* __restrict__ p, long n) {
  long i = (long)blockIdx.x * blockDim.x + threadIdx.x;
  if (i * 4 < n) {
    v4f z = {0.f, 0.f, 0.f, 0.f};
    ((v4f*)p)[i] = z;
  }
}

// ---------------------------------------------------------------------------
// xw = x @ W1^T + b1   (N x 64 -> N x 128), one wave per 16-row tile.
// ---------------------------------------------------------------------------
__global__ void k_xw(const float* __restrict__ x, const _Float16* __restrict__ w1s,
                     const float* __restrict__ b1, float* __restrict__ xw, int NT) {
  int wave = blockIdx.x * (blockDim.x >> 5) + (threadIdx.x >> 5);
  if (wave >= NT) return;                 // wave-uniform exit, EXEC stays all-ones
  int lane = threadIdx.x & 31;
  int grp = lane >> 4, ln = lane & 15;
  int rowBase = wave * 16;

  v8f acc[8];
#pragma unroll
  for (int n = 0; n < 8; ++n) {
    float bv = b1[n * 16 + ln];
#pragma unroll
    for (int r = 0; r < 8; ++r) acc[n][r] = bv;
  }

#pragma unroll
  for (int kt = 0; kt < INC / 32; ++kt) {
    v16h btile[8];
#pragma unroll
    for (int n = 0; n < 8; ++n)
      btile[n] = *(const v16h*)(w1s + (size_t)(((kt * 8) + n) * 32 + lane) * 16);
    v16h a = loadA_global_f32(x + (size_t)(rowBase + ln) * INC + kt * 32 + 8 * grp);
#pragma unroll
    for (int n = 0; n < 8; ++n) acc[n] = wmma_16x16x32(a, btile[n], acc[n]);
  }

#pragma unroll
  for (int n = 0; n < 8; ++n)
#pragma unroll
    for (int r = 0; r < 8; ++r)
      xw[(size_t)(rowBase + r + 8 * grp) * OUTC + n * 16 + ln] = acc[n][r];
}

// ---------------------------------------------------------------------------
// Edge scatter: agg[dst] += u[src]   (f32 global atomics, 4 ch per thread)
// ---------------------------------------------------------------------------
__global__ void k_scatter(const float* __restrict__ u, const int* __restrict__ srcI,
                          const int* __restrict__ dstI, float* __restrict__ agg, int E) {
  int idx = blockIdx.x * blockDim.x + threadIdx.x;
  if (idx >= E * 32) return;
  int e = idx >> 5;
  int c = (idx & 31) << 2;
  int s = srcI[e], d = dstI[e];
  v4f v = *(const v4f*)(u + (size_t)s * OUTC + c);
  float* o = agg + (size_t)d * OUTC + c;
  unsafeAtomicAdd(o + 0, v.x);
  unsafeAtomicAdd(o + 1, v.y);
  unsafeAtomicAdd(o + 2, v.z);
  unsafeAtomicAdd(o + 3, v.w);
}

// ---------------------------------------------------------------------------
// Fused MLP: h=relu(agg@m1^T+b1); h=relu(h@m2^T+b2); h=tanh(h@m3^T+b3);
//            u = relu(xw + h).  One wave per 16-row tile; LDS re-layout between
//            layers (wave-private slice -> only s_wait_dscnt needed).
// ---------------------------------------------------------------------------
__global__ void k_mlp(const float* __restrict__ agg,
                      const _Float16* __restrict__ m1s, const float* __restrict__ b1,
                      const _Float16* __restrict__ m2s, const float* __restrict__ b2,
                      const _Float16* __restrict__ m3s, const float* __restrict__ b3,
                      const float* __restrict__ xw, float* __restrict__ u, int NT) {
  __shared__ _Float16 hbuf[8 * 16 * OUTC];   // 4KB per wave, 32KB per 256-thread block
  int wave = blockIdx.x * (blockDim.x >> 5) + (threadIdx.x >> 5);
  if (wave >= NT) return;
  int lane = threadIdx.x & 31;
  int grp = lane >> 4, ln = lane & 15;
  int rowBase = wave * 16;
  _Float16* my = hbuf + (size_t)(threadIdx.x >> 5) * 16 * OUTC;

  // Pull upcoming weight images toward the WGP while layer 1 runs.
  __builtin_prefetch(m2s + (size_t)lane * 256, 0, 3);
  __builtin_prefetch(m3s + (size_t)lane * 256, 0, 3);

  v8f acc[8];

  // ---- layer 1: A from global f32 agg ----
#pragma unroll
  for (int n = 0; n < 8; ++n) {
    float bv = b1[n * 16 + ln];
#pragma unroll
    for (int r = 0; r < 8; ++r) acc[n][r] = bv;
  }
#pragma unroll
  for (int kt = 0; kt < 4; ++kt) {
    v16h btile[8];
#pragma unroll
    for (int n = 0; n < 8; ++n)
      btile[n] = *(const v16h*)(m1s + (size_t)(((kt * 8) + n) * 32 + lane) * 16);
    v16h a = loadA_global_f32(agg + (size_t)(rowBase + ln) * OUTC + kt * 32 + 8 * grp);
#pragma unroll
    for (int n = 0; n < 8; ++n) acc[n] = wmma_16x16x32(a, btile[n], acc[n]);
  }
#pragma unroll
  for (int n = 0; n < 8; ++n)
#pragma unroll
    for (int r = 0; r < 8; ++r) {
      float v = acc[n][r];
      v = v > 0.f ? v : 0.f;
      my[(r + 8 * grp) * OUTC + n * 16 + ln] = (_Float16)v;
    }
  asm volatile("s_wait_dscnt 0" ::: "memory");

  // ---- layer 2: A from LDS ----
#pragma unroll
  for (int n = 0; n < 8; ++n) {
    float bv = b2[n * 16 + ln];
#pragma unroll
    for (int r = 0; r < 8; ++r) acc[n][r] = bv;
  }
#pragma unroll
  for (int kt = 0; kt < 4; ++kt) {
    v16h btile[8];
#pragma unroll
    for (int n = 0; n < 8; ++n)
      btile[n] = *(const v16h*)(m2s + (size_t)(((kt * 8) + n) * 32 + lane) * 16);
    v16h a = loadA_lds_f16(my + ln * OUTC + kt * 32 + 8 * grp);
#pragma unroll
    for (int n = 0; n < 8; ++n) acc[n] = wmma_16x16x32(a, btile[n], acc[n]);
  }
  asm volatile("s_wait_dscnt 0" ::: "memory");   // all layer-2 A reads done before overwrite
#pragma unroll
  for (int n = 0; n < 8; ++n)
#pragma unroll
    for (int r = 0; r < 8; ++r) {
      float v = acc[n][r];
      v = v > 0.f ? v : 0.f;
      my[(r + 8 * grp) * OUTC + n * 16 + ln] = (_Float16)v;
    }
  asm volatile("s_wait_dscnt 0" ::: "memory");

  // ---- layer 3: A from LDS, tanh, fuse u = relu(xw + h) ----
#pragma unroll
  for (int n = 0; n < 8; ++n) {
    float bv = b3[n * 16 + ln];
#pragma unroll
    for (int r = 0; r < 8; ++r) acc[n][r] = bv;
  }
#pragma unroll
  for (int kt = 0; kt < 4; ++kt) {
    v16h btile[8];
#pragma unroll
    for (int n = 0; n < 8; ++n)
      btile[n] = *(const v16h*)(m3s + (size_t)(((kt * 8) + n) * 32 + lane) * 16);
    v16h a = loadA_lds_f16(my + ln * OUTC + kt * 32 + 8 * grp);
#pragma unroll
    for (int n = 0; n < 8; ++n) acc[n] = wmma_16x16x32(a, btile[n], acc[n]);
  }
#pragma unroll
  for (int n = 0; n < 8; ++n)
#pragma unroll
    for (int r = 0; r < 8; ++r) {
      float h = tanhf(acc[n][r]);
      size_t idx = (size_t)(rowBase + r + 8 * grp) * OUTC + n * 16 + ln;
      float v = xw[idx] + h;
      u[idx] = v > 0.f ? v : 0.f;
    }
}

// ---------------------------------------------------------------------------
// Column sum of u -> g[128]; per-block register accumulation + 1 atomic/col.
// ---------------------------------------------------------------------------
__global__ void k_colsum(const float* __restrict__ u, float* __restrict__ g, int N) {
  int col = threadIdx.x;            // 128 threads
  int row0 = blockIdx.x * 256;
  float s = 0.f;
  for (int r = 0; r < 256; ++r) {
    int row = row0 + r;
    if (row < N) s += u[(size_t)row * OUTC + col];
  }
  unsafeAtomicAdd(g + col, s);
}

__global__ void k_final(const float* __restrict__ g, const float* __restrict__ w2,
                        const float* __restrict__ b2, float* __restrict__ out) {
  int o = threadIdx.x;              // 128 threads
  float s = b2[o];
  for (int k = 0; k < OUTC; ++k) s += g[k] * w2[(size_t)o * OUTC + k];
  out[o] = s;
}

// ---------------------------------------------------------------------------
extern "C" void kernel_launch(void* const* d_in, const int* in_sizes, int n_in,
                              void* d_out, int out_size, void* d_ws, size_t ws_size,
                              hipStream_t stream) {
  const float* x   = (const float*)d_in[0];
  const float* u0  = (const float*)d_in[1];
  const int*   ei  = (const int*)d_in[2];
  const float* w1W = (const float*)d_in[3];
  const float* w1b = (const float*)d_in[4];
  const float* m1W = (const float*)d_in[5];
  const float* m1b = (const float*)d_in[6];
  const float* m2W = (const float*)d_in[7];
  const float* m2b = (const float*)d_in[8];
  const float* m3W = (const float*)d_in[9];
  const float* m3b = (const float*)d_in[10];
  const float* w2W = (const float*)d_in[11];
  const float* w2b = (const float*)d_in[12];

  int N = in_sizes[0] / INC;      // 50000
  int E = in_sizes[2] / 2;        // 800000
  int NT = (N + 15) / 16;         // 3125 (exact)

  char* ws = (char*)d_ws;
  size_t nodeBytes = (size_t)N * OUTC * sizeof(float);
  float*    xw  = (float*)ws;                       size_t off = nodeBytes;
  float*    u   = (float*)(ws + off);               off += nodeBytes;
  float*    agg = (float*)(ws + off);               off += nodeBytes;
  _Float16* w1s = (_Float16*)(ws + off);            off += (size_t)2 * 8 * 32 * 16 * 2;
  _Float16* m1s = (_Float16*)(ws + off);            off += (size_t)4 * 8 * 32 * 16 * 2;
  _Float16* m2s = (_Float16*)(ws + off);            off += (size_t)4 * 8 * 32 * 16 * 2;
  _Float16* m3s = (_Float16*)(ws + off);            off += (size_t)4 * 8 * 32 * 16 * 2;
  float*    g   = (float*)(ws + off);

  const int* srcI = ei;
  const int* dstI = ei + E;

  // weight pre-swizzle (f32 -> f16 B-tile layout)
  k_swz<<<(2 * 8 * 32 * 16 + 255) / 256, 256, 0, stream>>>(w1W, w1s, INC);
  k_swz<<<(4 * 8 * 32 * 16 + 255) / 256, 256, 0, stream>>>(m1W, m1s, OUTC);
  k_swz<<<(4 * 8 * 32 * 16 + 255) / 256, 256, 0, stream>>>(m2W, m2s, OUTC);
  k_swz<<<(4 * 8 * 32 * 16 + 255) / 256, 256, 0, stream>>>(m3W, m3s, OUTC);

  hipMemcpyAsync(u, u0, nodeBytes, hipMemcpyDeviceToDevice, stream);

  k_xw<<<(NT + 7) / 8, 256, 0, stream>>>(x, w1s, w1b, xw, NT);

  long nElems = (long)N * OUTC;
  int zeroBlocks = (int)((nElems / 4 + 255) / 256);
  int scatBlocks = (int)(((long)E * 32 + 255) / 256);
  for (int t = 0; t < 4; ++t) {
    k_zero<<<zeroBlocks, 256, 0, stream>>>(agg, nElems);
    k_scatter<<<scatBlocks, 256, 0, stream>>>(u, srcI, dstI, agg, E);
    k_mlp<<<(NT + 7) / 8, 256, 0, stream>>>(agg, m1s, m1b, m2s, m2b, m3s, m3b, xw, u, NT);
  }

  k_zero<<<1, 64, 0, stream>>>(g, OUTC);
  k_colsum<<<(N + 255) / 256, 128, 0, stream>>>(u, g, N);
  k_final<<<1, 128, 0, stream>>>(g, w2W, w2b, (float*)d_out);
}
